// InferenceEngine_50268297232824
// MI455X (gfx1250) — matrix operand order repaired and checked by reading, not compiled, not run on hardware
//
#include <hip/hip_runtime.h>

typedef __attribute__((ext_vector_type(16))) _Float16 v16h;
typedef __attribute__((ext_vector_type(8)))  float    v8f;

#define KR 16      // rules / facts
#define DP 64      // feature dim
#define MS 32      // scratchpad slots
#define LDC 68     // padded LDS row stride (floats)
#define LDM 36     // padded stride for match_scores rows (>=32 for zero-padded K)
#define CEPS 1e-8f

// ---- WMMA fragment loaders (16x16x32 f16, contiguous K order in A and B, not the ISA table order) ----
// A 16x32 f16 per lane: one contiguous span of 16 values: K = hi*16 + 0..15
__device__ inline v16h load_fragA32(const float* __restrict__ p, int hi) {
  v16h f;
  const float* q0 = p + hi * 16;
#pragma unroll
  for (int e = 0; e < 16; ++e) f[e] = (_Float16)q0[e];
  return f;
}
// B 32x16 f16 per lane: one contiguous span of 16 values: K = hi*16 + 0..15
__device__ inline v16h load_fragB32(const float* __restrict__ p, int hi) {
  v16h f;
  const float* q = p + hi * 16;
#pragma unroll
  for (int e = 0; e < 16; ++e) f[e] = (_Float16)q[e];
  return f;
}
// B 32x16 f16 with strided K (column read from row-major LDS): K index = hi*16+e
__device__ inline v16h load_fragB_col(const float* __restrict__ p, int ld, int hi) {
  v16h f;
  const float* q = p + hi * 16 * ld;
#pragma unroll
  for (int e = 0; e < 16; ++e) f[e] = (_Float16)q[e * ld];
  return f;
}

__device__ inline v8f wmma_f16(v16h a, v16h b, v8f c) {
  return __builtin_amdgcn_wmma_f32_16x16x32_f16(false, a, false, b,
                                                (short)0, c, false, false);
}

__device__ inline float sigf(float x) { return 1.0f / (1.0f + __expf(-x)); }

__global__ __launch_bounds__(32)
void InferenceEngine_kernel(const float* __restrict__ content,
                            const float* __restrict__ types,
                            const float* __restrict__ conf,
                            const float* __restrict__ Sin,
                            const float* __restrict__ W_ante,
                            const float* __restrict__ W_cons,
                            const float* __restrict__ W_gw,
                            const float* __restrict__ W_gb,
                            const float* __restrict__ logthr,
                            float* __restrict__ out_S,
                            float* __restrict__ out_conf) {
  // sC has 32 rows: rows 16..31 are zero (K-padding for the 16->32 padded WMMA)
  __shared__ float sC[2 * KR * LDC];  // content [16][64] + zero pad rows
  __shared__ float sA[KR * LDC];      // ante, later matched
  __shared__ float sNF[KR * LDC];     // new_fact
  __shared__ float sS[MS * LDC];      // scratchpad S [32][64]
  __shared__ float sM[KR * LDM];      // match_scores [16][16], cols 16..31 zero
  __shared__ float sCn[KR], sAn[KR], sCf[KR], sPf[KR], sPr[KR];
  __shared__ float sFire[KR], sDC[KR];
  __shared__ float sG[DP];

  const int b = blockIdx.x;
  const int lane = threadIdx.x;
  const int hi = lane >> 4, lo = lane & 15;

  // ---- stage content and S into LDS (coalesced), zero pad regions ----
  const float* cb = content + (size_t)b * KR * DP;
  for (int t = lane; t < KR * DP; t += 32) sC[(t >> 6) * LDC + (t & 63)] = cb[t];
  for (int t = lane; t < KR * LDC; t += 32) sC[KR * LDC + t] = 0.0f;  // zero rows 16..31
  for (int t = lane; t < KR * LDM; t += 32) sM[t] = 0.0f;             // zero match_scores pad
  const float* sb = Sin + (size_t)b * MS * DP;
  for (int t = lane; t < MS * DP; t += 32) sS[(t >> 6) * LDC + (t & 63)] = sb[t];
  __syncthreads();

  if (lane < KR) {
    float ss = 0.f;
    for (int d = 0; d < DP; ++d) { float v = sC[lane * LDC + d]; ss += v * v; }
    sCn[lane] = fmaxf(sqrtf(ss), CEPS);
    sCf[lane] = conf[(size_t)b * KR + lane];
    sPf[lane] = types[((size_t)b * KR + lane) * 3 + 0];
    sPr[lane] = types[((size_t)b * KR + lane) * 3 + 1];
  }
  __syncthreads();

  // ---- Phase 1: ante = content @ W_ante^T  (8 WMMA) ----
  {
    v16h a0 = load_fragA32(sC + lo * LDC + 0, hi);
    v16h a1 = load_fragA32(sC + lo * LDC + 32, hi);
#pragma unroll
    for (int n0 = 0; n0 < DP; n0 += 16) {
      int n = n0 + lo;
      v8f acc = {};
      acc = wmma_f16(a0, load_fragB32(W_ante + n * DP + 0, hi), acc);
      acc = wmma_f16(a1, load_fragB32(W_ante + n * DP + 32, hi), acc);
#pragma unroll
      for (int r = 0; r < 8; ++r) sA[(r + 8 * hi) * LDC + n] = acc[r];
    }
  }
  __syncthreads();
  if (lane < KR) {
    float ss = 0.f;
    for (int d = 0; d < DP; ++d) { float v = sA[lane * LDC + d]; ss += v * v; }
    sAn[lane] = fmaxf(sqrtf(ss), CEPS);
  }
  __syncthreads();

  // ---- Phase 2: dots = ante @ content^T, scale -> match_scores (2 WMMA) ----
  {
    v8f acc = {};
    acc = wmma_f16(load_fragA32(sA + lo * LDC + 0, hi),
                   load_fragB32(sC + lo * LDC + 0, hi), acc);
    acc = wmma_f16(load_fragA32(sA + lo * LDC + 32, hi),
                   load_fragB32(sC + lo * LDC + 32, hi), acc);
    float scl = sPf[lo] / sCn[lo];
#pragma unroll
    for (int r = 0; r < 8; ++r) {
      int m = r + 8 * hi;
      sM[m * LDM + lo] = acc[r] / sAn[m] * scl;
    }
  }
  __syncthreads();

  // match_strength -> fire -> derived_conf
  if (lane < KR) {
    float s = 0.f;
    for (int j = 0; j < KR; ++j) s += sM[lane * LDM + j] * sCf[j];
    float thr = __expf(logthr[0]);
    float fire = sigf((s - thr) * 2.0f) * sPr[lane];
    sFire[lane] = fire;
    sDC[lane] = fire * sCf[lane];
  }
  __syncthreads();

  // ---- Phase 3: matched = match_scores @ content (4 WMMA, K zero-padded) ----
  {
    v16h am = load_fragA32(sM + lo * LDM, hi);  // cols 16..31 are zero
#pragma unroll
    for (int n0 = 0; n0 < DP; n0 += 16) {
      int n = n0 + lo;
      v8f acc = {};
      // B[k][n] = content[k][n]; rows 16..31 of sC are zero
      acc = wmma_f16(am, load_fragB_col(sC + n, LDC, hi), acc);
#pragma unroll
      for (int r = 0; r < 8; ++r) sA[(r + 8 * hi) * LDC + n] = acc[r];  // overwrite ante
    }
  }
  __syncthreads();

  // ---- Phase 4: new_fact = [content||matched] @ W_cons^T (16 WMMA) ----
  {
    v16h x0 = load_fragA32(sC + lo * LDC + 0, hi);
    v16h x1 = load_fragA32(sC + lo * LDC + 32, hi);
    v16h x2 = load_fragA32(sA + lo * LDC + 0, hi);
    v16h x3 = load_fragA32(sA + lo * LDC + 32, hi);
#pragma unroll
    for (int n0 = 0; n0 < DP; n0 += 16) {
      int n = n0 + lo;
      const float* wr = W_cons + n * (2 * DP);
      v8f acc = {};
      acc = wmma_f16(x0, load_fragB32(wr + 0, hi), acc);
      acc = wmma_f16(x1, load_fragB32(wr + 32, hi), acc);
      acc = wmma_f16(x2, load_fragB32(wr + 64, hi), acc);
      acc = wmma_f16(x3, load_fragB32(wr + 96, hi), acc);
#pragma unroll
      for (int r = 0; r < 8; ++r) sNF[(r + 8 * hi) * LDC + n] = acc[r];
    }
  }
  __syncthreads();

  // ---- Phase 5: sequential soft-write scan (lane = S row, wave32) ----
  const int m = lane;
  for (int k = 0; k < KR; ++k) {
    float fire = sFire[k];  // uniform across wave
    if (fire > 0.1f) {
      float dconf = sDC[k];
      const float* cand = sNF + k * LDC;
      float dot = 0.f, ssq = 0.f, csq = 0.f;
      for (int d = 0; d < DP; ++d) {
        float sv = sS[m * LDC + d], cv = cand[d];
        dot += sv * cv; ssq += sv * sv; csq += cv * cv;
      }
      float sc = dot / (fmaxf(sqrtf(csq), CEPS) * fmaxf(sqrtf(ssq), CEPS));
      // softmax(4*sc) over 32 lanes
      float mx = sc;
#pragma unroll
      for (int off = 16; off > 0; off >>= 1) mx = fmaxf(mx, __shfl_xor(mx, off, 32));
      float e = __expf((sc - mx) * 4.0f);
      float se = e;
#pragma unroll
      for (int off = 16; off > 0; off >>= 1) se += __shfl_xor(se, off, 32);
      float alpha = e / se;
      // argmax (first index on ties)
      float bv = sc; int bi = m;
#pragma unroll
      for (int off = 16; off > 0; off >>= 1) {
        float ov = __shfl_xor(bv, off, 32);
        int oi = __shfl_xor(bi, off, 32);
        if (ov > bv || (ov == bv && oi < bi)) { bv = ov; bi = oi; }
      }
      int best = bi;
      // gamma = sigmoid(W_gw @ [best_slot, cand] + b): 2 outputs/lane
      float acc0 = W_gb[lane], acc1 = W_gb[lane + 32];
      const float* w0 = W_gw + lane * (2 * DP);
      const float* w1 = W_gw + (lane + 32) * (2 * DP);
      for (int i = 0; i < DP; ++i) {
        float bs = sS[best * LDC + i];
        acc0 += w0[i] * bs; acc1 += w1[i] * bs;
      }
      for (int i = 0; i < DP; ++i) {
        float cv = cand[i];
        acc0 += w0[DP + i] * cv; acc1 += w1[DP + i] * cv;
      }
      sG[lane] = sigf(acc0);
      sG[lane + 32] = sigf(acc1);
      __syncthreads();  // single-wave WG: barrier is a NOP; LDS ops stay in order
      // S[m] += dconf * alpha_m * (new_content - S[m])
      for (int d = 0; d < DP; ++d) {
        float bs = sS[best * LDC + d];  // read-before-write: DS in order
        float cv = cand[d];
        float g = sG[d];
        float nc = g * cv + (1.0f - g) * bs;
        float sv = sS[m * LDC + d];
        sS[m * LDC + d] = sv + dconf * alpha * (nc - sv);
      }
      __syncthreads();
    }
  }
  __syncthreads();

  // ---- outputs ----
  if (lane == 0) {
    float bd = sDC[0];
    for (int j = 1; j < KR; ++j) bd = fmaxf(bd, sDC[j]);
    out_conf[b] = bd;
  }
  float* ob = out_S + (size_t)b * MS * DP;
  for (int t = lane; t < MS * DP; t += 32) ob[t] = sS[(t >> 6) * LDC + (t & 63)];
}

extern "C" void kernel_launch(void* const* d_in, const int* in_sizes, int n_in,
                              void* d_out, int out_size, void* d_ws, size_t ws_size,
                              hipStream_t stream) {
  const float* content = (const float*)d_in[0];
  const float* types   = (const float*)d_in[1];
  const float* conf    = (const float*)d_in[2];
  const float* Sin     = (const float*)d_in[3];
  const float* W_ante  = (const float*)d_in[4];
  const float* W_cons  = (const float*)d_in[5];
  const float* W_gw    = (const float*)d_in[6];
  const float* W_gb    = (const float*)d_in[7];
  const float* logthr  = (const float*)d_in[8];

  const int B = in_sizes[0] / (KR * DP);  // content is [B,16,64]
  float* out_S = (float*)d_out;
  float* out_conf = out_S + (size_t)B * MS * DP;

  InferenceEngine_kernel<<<B, 32, 0, stream>>>(
      content, types, conf, Sin, W_ante, W_cons, W_gw, W_gb, logthr,
      out_S, out_conf);
}
